// MHAP_5033701671185
// MI455X (gfx1250) — matrix operand order repaired and checked by reading, not compiled, not run on hardware
//
#include <hip/hip_runtime.h>
#include <hip/hip_bf16.h>

#define N_TOK 16384
#define DIMV  1024
#define HEADS 16
#define HDIM  64
#define MID   24
#define TOPK  8192
#define NCH1  64   // split-K chunks for conv1
#define ACH   64   // n-chunks for attention accumulation

typedef __attribute__((ext_vector_type(16))) __bf16 v16bf;
typedef __attribute__((ext_vector_type(8)))  float  v8f;

union BF16x16 {
  v16bf v;
  uint4 q[2];
};

__device__ __forceinline__ unsigned short f2bf(float f) {
  unsigned int u = __float_as_uint(f);
  unsigned int r = u + 0x7fffu + ((u >> 16) & 1u);   // round-to-nearest-even
  return (unsigned short)(r >> 16);
}

// -------- split fp32 -> bf16 hi + bf16 lo (residual) --------
__global__ void split_bf16(const float* __restrict__ src,
                           unsigned short* __restrict__ hi,
                           unsigned short* __restrict__ lo, int n) {
  int i = blockIdx.x * blockDim.x + threadIdx.x;
  int stride = gridDim.x * blockDim.x;
  for (; i < n; i += stride) {
    float f = src[i];
    unsigned short h = f2bf(f);
    float fh = __uint_as_float(((unsigned int)h) << 16);
    unsigned short l = f2bf(f - fh);
    hi[i] = h;
    lo[i] = l;
  }
}

// -------- QKV GEMM: C[n,o] = sum_c x[n,c]*W[o,c] + b[o], via bf16x3 WMMA --------
__global__ void __launch_bounds__(256) gemm_qkv(
    const unsigned short* __restrict__ xhi, const unsigned short* __restrict__ xlo,
    const unsigned short* __restrict__ whi, const unsigned short* __restrict__ wlo,
    const float* __restrict__ bq, const float* __restrict__ bk,
    const float* __restrict__ bv, float* __restrict__ out) {
  const int z = blockIdx.z;
  const unsigned short* Wh = whi + (size_t)z * DIMV * DIMV;
  const unsigned short* Wl = wlo + (size_t)z * DIMV * DIMV;
  const float* bias = (z == 0) ? bq : (z == 1) ? bk : bv;
  float* C = out + (size_t)z * N_TOK * DIMV;

  const int tid   = threadIdx.x;
  const int lane  = tid & 31;
  const int wid   = tid >> 5;       // 0..7
  const int l16   = lane & 15;
  const int g     = lane >> 4;      // 0/1: lane half-group
  const int waveM = wid & 1;        // 2 waves over M
  const int waveN = wid >> 1;       // 4 waves over N
  const int m0 = blockIdx.x * 64  + waveM * 32;
  const int n0 = blockIdx.y * 128 + waveN * 32;

  v8f zero = {0.f, 0.f, 0.f, 0.f, 0.f, 0.f, 0.f, 0.f};
  v8f acc[2][2];
  acc[0][0] = zero; acc[0][1] = zero; acc[1][0] = zero; acc[1][1] = zero;

  for (int kk = 0; kk < DIMV; kk += 32) {
    BF16x16 ah[2], al[2], bh[2], bl[2];
    // A fragments: 16x32 bf16; lane l16 holds row m, halves 0..7 = K kk+8g..,
    // halves 8..15 = K kk+16+8g.. (per CDNA5 ISA A layout)
    #pragma unroll
    for (int i = 0; i < 2; ++i) {
      const size_t row = (size_t)(m0 + i * 16 + l16) * DIMV;
      const unsigned short* ph = xhi + row + kk;
      const unsigned short* pl = xlo + row + kk;
      ah[i].q[0] = *(const uint4*)(ph + g * 8);
      ah[i].q[1] = *(const uint4*)(ph + 16 + g * 8);
      al[i].q[0] = *(const uint4*)(pl + g * 8);
      al[i].q[1] = *(const uint4*)(pl + 16 + g * 8);
    }
    // B fragments: 32x16 bf16; lane holds column o = n0+l16, halves 0..7 = K kk+8g..,
    // halves 8..15 = K kk+16+8g.. of W[o, :] (row-major [O,C])
    #pragma unroll
    for (int j = 0; j < 2; ++j) {
      const size_t row = (size_t)(n0 + j * 16 + l16) * DIMV;
      const unsigned short* ph = Wh + row + kk + g * 8;
      const unsigned short* pl = Wl + row + kk + g * 8;
      bh[j].q[0] = *(const uint4*)(ph);
      bh[j].q[1] = *(const uint4*)(ph + 16);
      bl[j].q[0] = *(const uint4*)(pl);
      bl[j].q[1] = *(const uint4*)(pl + 16);
    }
    #pragma unroll
    for (int i = 0; i < 2; ++i) {
      #pragma unroll
      for (int j = 0; j < 2; ++j) {
        acc[i][j] = __builtin_amdgcn_wmma_f32_16x16x32_bf16(
            false, ah[i].v, false, bh[j].v, (short)0, acc[i][j], false, false);
        acc[i][j] = __builtin_amdgcn_wmma_f32_16x16x32_bf16(
            false, ah[i].v, false, bl[j].v, (short)0, acc[i][j], false, false);
        acc[i][j] = __builtin_amdgcn_wmma_f32_16x16x32_bf16(
            false, al[i].v, false, bh[j].v, (short)0, acc[i][j], false, false);
      }
    }
  }
  // Epilogue: C/D layout — lane writes rows m0+8g+v, column n0+l16
  #pragma unroll
  for (int i = 0; i < 2; ++i) {
    #pragma unroll
    for (int j = 0; j < 2; ++j) {
      const int col = n0 + j * 16 + l16;
      const float b = bias[col];
      #pragma unroll
      for (int v = 0; v < 8; ++v) {
        const int row = m0 + i * 16 + g * 8 + v;
        C[(size_t)row * DIMV + col] = acc[i][j][v] + b;
      }
    }
  }
}

// -------- conv1 split-K partials: part[c][m][l] = sum_{n in chunk c} W1[m,n]*Q[n,l] --------
__global__ void __launch_bounds__(256) conv1_partial(const float* __restrict__ Qf,
                                                     const float* __restrict__ W1,
                                                     float* __restrict__ part) {
  const int l = blockIdx.x * 256 + threadIdx.x;   // 0..1023
  const int c = blockIdx.y;                        // 0..NCH1-1
  float acc[MID];
  #pragma unroll
  for (int m = 0; m < MID; ++m) acc[m] = 0.f;
  const int nbeg = c * (N_TOK / NCH1);
  for (int n = nbeg; n < nbeg + N_TOK / NCH1; ++n) {
    float q = Qf[(size_t)n * DIMV + l];
    #pragma unroll
    for (int m = 0; m < MID; ++m) acc[m] += W1[(size_t)m * N_TOK + n] * q;
  }
  #pragma unroll
  for (int m = 0; m < MID; ++m)
    part[((size_t)c * MID + m) * DIMV + l] = acc[m];
}

// -------- conv finalize: bias + relu + W2 reduction -> qvec[1024] --------
__global__ void __launch_bounds__(256) conv_finalize(const float* __restrict__ part,
                                                     const float* __restrict__ b1,
                                                     const float* __restrict__ W2,
                                                     const float* __restrict__ b2,
                                                     float* __restrict__ qvec) {
  const int l = blockIdx.x * 256 + threadIdx.x;
  if (l >= DIMV) return;
  float o = b2[0];
  #pragma unroll
  for (int m = 0; m < MID; ++m) {
    float t = b1[m];
    for (int c = 0; c < NCH1; ++c) t += part[((size_t)c * MID + m) * DIMV + l];
    t = fmaxf(t, 0.f);
    o += W2[m] * t;
  }
  qvec[l] = o;
}

// -------- attention logits: sc[h][n] = 0.125 * dot(qvec[h*64:], K[n, h*64:]) --------
__global__ void __launch_bounds__(256) scores_kernel(const float* __restrict__ Kf,
                                                     const float* __restrict__ qvec,
                                                     float* __restrict__ sc) {
  __shared__ float red[256];
  const int n = blockIdx.x;
  const int tid = threadIdx.x;
  const float4 kv = *(const float4*)(Kf + (size_t)n * DIMV + tid * 4);
  const float4 qv = *(const float4*)(qvec + tid * 4);
  red[tid] = kv.x * qv.x + kv.y * qv.y + kv.z * qv.z + kv.w * qv.w;
  __syncthreads();
  if (tid < HEADS) {
    float s = 0.f;
    #pragma unroll
    for (int i = 0; i < 16; ++i) s += red[tid * 16 + i];
    sc[(size_t)tid * N_TOK + n] = s * 0.125f;
  }
}

// -------- per-head: max, exact top-k threshold (radix select), masked denom --------
__global__ void __launch_bounds__(1024) select_kernel(const float* __restrict__ sc,
                                                      float* __restrict__ stats) {
  const int h = blockIdx.x;
  const int tid = threadIdx.x;
  const float* s = sc + (size_t)h * N_TOK;
  __shared__ float redf[1024];
  __shared__ unsigned int cnt;

  // block max
  float mx = -3.0e38f;
  for (int n = tid; n < N_TOK; n += 1024) mx = fmaxf(mx, s[n]);
  redf[tid] = mx;
  __syncthreads();
  for (int off = 512; off > 0; off >>= 1) {
    if (tid < off) redf[tid] = fmaxf(redf[tid], redf[tid + off]);
    __syncthreads();
  }
  mx = redf[0];
  __syncthreads();

  // radix select of TOPK-th largest on the monotone uint mapping
  unsigned int best = 0;
  for (int bit = 31; bit >= 0; --bit) {
    unsigned int cand = best | (1u << bit);
    if (tid == 0) cnt = 0;
    __syncthreads();
    unsigned int local = 0;
    for (int n = tid; n < N_TOK; n += 1024) {
      unsigned int b = __float_as_uint(s[n]);
      unsigned int u = (b & 0x80000000u) ? ~b : (b | 0x80000000u);
      local += (u >= cand) ? 1u : 0u;
    }
    atomicAdd(&cnt, local);
    __syncthreads();
    if (cnt >= TOPK) best = cand;
    __syncthreads();
  }

  // masked softmax denominator over kept entries
  float dsum = 0.f;
  for (int n = tid; n < N_TOK; n += 1024) {
    float v = s[n];
    unsigned int b = __float_as_uint(v);
    unsigned int u = (b & 0x80000000u) ? ~b : (b | 0x80000000u);
    if (u >= best) dsum += __expf(v - mx);
  }
  redf[tid] = dsum;
  __syncthreads();
  for (int off = 512; off > 0; off >>= 1) {
    if (tid < off) redf[tid] += redf[tid + off];
    __syncthreads();
  }
  if (tid == 0) {
    stats[h * 4 + 0] = mx;
    stats[h * 4 + 1] = redf[0];
    stats[h * 4 + 2] = __uint_as_float(best);   // threshold in uint-mapped space
  }
}

// -------- masked A@V partials: apart[h][c][d] --------
__global__ void __launch_bounds__(256) attn_partial(const float* __restrict__ sc,
                                                    const float* __restrict__ Vf,
                                                    const float* __restrict__ stats,
                                                    float* __restrict__ apart) {
  const int c = blockIdx.x;     // 0..ACH-1
  const int h = blockIdx.y;     // 0..15
  const int tid = threadIdx.x;
  const int d = tid & 63;
  const int ns = tid >> 6;      // 0..3
  const float mx = stats[h * 4 + 0];
  const float inv = 1.0f / stats[h * 4 + 1];
  const unsigned int thr = __float_as_uint(stats[h * 4 + 2]);
  float acc = 0.f;
  const int nbeg = c * (N_TOK / ACH);
  for (int n = nbeg + ns; n < nbeg + N_TOK / ACH; n += 4) {
    float v = sc[(size_t)h * N_TOK + n];
    unsigned int b = __float_as_uint(v);
    unsigned int u = (b & 0x80000000u) ? ~b : (b | 0x80000000u);
    if (u >= thr) {
      float w = __expf(v - mx) * inv;
      acc += w * Vf[(size_t)n * DIMV + h * HDIM + d];
    }
  }
  __shared__ float sm[256];
  sm[tid] = acc;
  __syncthreads();
  if (ns == 0) {
    float t = sm[d] + sm[64 + d] + sm[128 + d] + sm[192 + d];
    apart[((size_t)h * ACH + c) * HDIM + d] = t;
  }
}

// -------- final fixed-order reduction into d_out[1024] --------
__global__ void __launch_bounds__(256) attn_reduce(const float* __restrict__ apart,
                                                   float* __restrict__ out) {
  const int t = blockIdx.x * 256 + threadIdx.x;
  if (t >= DIMV) return;
  const int h = t >> 6, d = t & 63;
  float s = 0.f;
  for (int c = 0; c < ACH; ++c) s += apart[((size_t)h * ACH + c) * HDIM + d];
  out[t] = s;
}

extern "C" void kernel_launch(void* const* d_in, const int* in_sizes, int n_in,
                              void* d_out, int out_size, void* d_ws, size_t ws_size,
                              hipStream_t stream) {
  (void)in_sizes; (void)n_in; (void)out_size; (void)ws_size;
  const float* x  = (const float*)d_in[0];
  const float* Wq = (const float*)d_in[1];
  const float* bq = (const float*)d_in[2];
  const float* Wk = (const float*)d_in[3];
  const float* bk = (const float*)d_in[4];
  const float* Wv = (const float*)d_in[5];
  const float* bv = (const float*)d_in[6];
  const float* W1 = (const float*)d_in[7];
  const float* b1 = (const float*)d_in[8];
  const float* W2 = (const float*)d_in[9];
  const float* b2 = (const float*)d_in[10];

  char* ws = (char*)d_ws;
  size_t off = 0;
  float* Qf = (float*)(ws + off); off += (size_t)N_TOK * DIMV * 4;   // Q fp32
  float* Kf = (float*)(ws + off); off += (size_t)N_TOK * DIMV * 4;   // K fp32
  float* Vf = (float*)(ws + off); off += (size_t)N_TOK * DIMV * 4;   // V fp32
  unsigned short* xhi = (unsigned short*)(ws + off); off += (size_t)N_TOK * DIMV * 2;
  unsigned short* xlo = (unsigned short*)(ws + off); off += (size_t)N_TOK * DIMV * 2;
  unsigned short* whi = (unsigned short*)(ws + off); off += (size_t)3 * DIMV * DIMV * 2;
  unsigned short* wlo = (unsigned short*)(ws + off); off += (size_t)3 * DIMV * DIMV * 2;
  float* part1 = (float*)(ws + off); off += (size_t)NCH1 * MID * DIMV * 4;
  float* qvec  = (float*)(ws + off); off += (size_t)DIMV * 4;
  float* sc    = (float*)(ws + off); off += (size_t)HEADS * N_TOK * 4;
  float* stats = (float*)(ws + off); off += (size_t)HEADS * 4 * 4;
  float* apart = (float*)(ws + off); off += (size_t)HEADS * ACH * HDIM * 4;

  // 1) fp32 -> bf16 hi/lo splits
  split_bf16<<<2048, 256, 0, stream>>>(x, xhi, xlo, N_TOK * DIMV);
  split_bf16<<<512, 256, 0, stream>>>(Wq, whi, wlo, DIMV * DIMV);
  split_bf16<<<512, 256, 0, stream>>>(Wk, whi + (size_t)DIMV * DIMV,
                                      wlo + (size_t)DIMV * DIMV, DIMV * DIMV);
  split_bf16<<<512, 256, 0, stream>>>(Wv, whi + (size_t)2 * DIMV * DIMV,
                                      wlo + (size_t)2 * DIMV * DIMV, DIMV * DIMV);
  // 2) Q/K/V GEMMs (WMMA bf16x3)
  gemm_qkv<<<dim3(N_TOK / 64, DIMV / 128, 3), 256, 0, stream>>>(
      xhi, xlo, whi, wlo, bq, bk, bv, Qf);
  // 3) conv pipeline -> qvec
  conv1_partial<<<dim3(DIMV / 256, NCH1), 256, 0, stream>>>(Qf, W1, part1);
  conv_finalize<<<DIMV / 256, 256, 0, stream>>>(part1, b1, W2, b2, qvec);
  // 4) logits
  scores_kernel<<<N_TOK, 256, 0, stream>>>(Kf, qvec, sc);
  // 5) per-head max / top-k threshold / masked denom
  select_kernel<<<HEADS, 1024, 0, stream>>>(sc, stats);
  // 6) masked weighted sum over V
  attn_partial<<<dim3(ACH, HEADS), 256, 0, stream>>>(sc, Vf, stats, apart);
  attn_reduce<<<DIMV / 256, 256, 0, stream>>>(apart, (float*)d_out);
}
